// EmbeddingOutputDecoder_33543694581830
// MI455X (gfx1250) — matrix operand order repaired and checked by reading, not compiled, not run on hardware
//
#include <hip/hip_runtime.h>
#include <hip/hip_bf16.h>
#include <math.h>

// ---------------------------------------------------------------------------
// Problem constants
// ---------------------------------------------------------------------------
#define D_IN 512
#define HDIM 1024
#define VOC  32000
#define SLEN 64
#define BSZ  128
#define TLEN 48
#define MARGIN 0.1f

typedef __attribute__((ext_vector_type(16))) _Float16 v16h;
typedef __attribute__((ext_vector_type(8)))  _Float16 v8h;
typedef __attribute__((ext_vector_type(8)))  float    v8f;

// ---------------------------------------------------------------------------
// WMMA fragment loaders (wave32; A and B use contiguous k = 16*(lane/16) + s)
// ---------------------------------------------------------------------------
// A-matrix 16x32 f16: lane&15 = row; lanes 0-15 hold K=0..15;
// lanes 16-31 hold K=16..31.  A stored row-major (M x K) f16:
// two contiguous 16-byte loads per lane, no conversions in the K-loop.
__device__ __forceinline__ v16h load_a_frag_f16(const _Float16* __restrict__ A,
                                                int K, int rowBase, int kb, int lane) {
    int row  = rowBase + (lane & 15);
    int koff = (lane < 16) ? 0 : 16;
    const _Float16* p = A + (size_t)row * K + kb + koff;
    v8h lo = *reinterpret_cast<const v8h*>(p);
    v8h hi = *reinterpret_cast<const v8h*>(p + 8);
    v16h a;
#pragma unroll
    for (int i = 0; i < 8; ++i) {
        a[i]     = lo[i];
        a[i + 8] = hi[i];
    }
    return a;
}

// B-matrix 32x16 f16: lane&15 = col (N); lanes 0-15 hold K=0..15,
// lanes 16-31 hold K=16..31. Weights stored row-major (N x K) f16, so this is
// one contiguous 32-byte load per lane.
__device__ __forceinline__ v16h load_b_frag_f16(const _Float16* __restrict__ W,
                                                int K, int colBase, int kb, int lane) {
    int col = colBase + (lane & 15);
    int kB  = (lane < 16) ? 0 : 16;
    return *reinterpret_cast<const v16h*>(W + (size_t)col * K + kb + kB);
}

__device__ __forceinline__ float sigf(float x) { return 1.0f / (1.0f + expf(-x)); }

// monotonic unsigned key for float atomic max (handles negatives)
__device__ __forceinline__ unsigned fkey(float f) {
    unsigned u = __float_as_uint(f);
    return (u & 0x80000000u) ? ~u : (u | 0x80000000u);
}
__device__ __forceinline__ float funkey(unsigned k) {
    unsigned u = (k & 0x80000000u) ? (k & 0x7fffffffu) : ~k;
    return __uint_as_float(u);
}

// ---------------------------------------------------------------------------
// Generic WMMA GEMM:  C[M x N] = A[M x K](f16) @ W[N x K](f16)^T  (f32 accum)
// grid = (N/64, M/16), block = 128 (4 waves, each owns one 16-col tile)
// ---------------------------------------------------------------------------
__global__ void k_gemm(const _Float16* __restrict__ A, const _Float16* __restrict__ W,
                       float* __restrict__ C, int M, int K, int N) {
    int lane  = threadIdx.x & 31;
    int wave  = threadIdx.x >> 5;
    int ntile = blockIdx.x * 4 + wave;
    int mtile = blockIdx.y;
    int colBase = ntile * 16;
    int rowBase = mtile * 16;

    v8f acc = {};
    for (int kb = 0; kb < K; kb += 32) {
        v16h a = load_a_frag_f16(A, K, rowBase, kb, lane);
        v16h b = load_b_frag_f16(W, K, colBase, kb, lane);
        acc = __builtin_amdgcn_wmma_f32_16x16x32_f16(false, a, false, b,
                                                     (short)0, acc, false, false);
    }
    int col = colBase + (lane & 15);
    int rof = (lane < 16) ? 0 : 8;
#pragma unroll
    for (int r = 0; r < 8; ++r)
        C[(size_t)(rowBase + r + rof) * N + col] = acc[r];
}

// ---------------------------------------------------------------------------
// Fused vocab kernel: rowwise max_{j != tgt} of logit @ Woutn^T
// M=128, K=512, N=32000.  grid = 500, block = 128 (4 waves x 1 ntile each)
// ---------------------------------------------------------------------------
__global__ void k_vocab_max(const _Float16* __restrict__ logit,
                            const _Float16* __restrict__ Wn,
                            const int* __restrict__ y,
                            unsigned* __restrict__ mmax_key, int t) {
    const int K = D_IN;
    int lane  = threadIdx.x & 31;
    int wave  = threadIdx.x >> 5;
    int ntile = blockIdx.x * 4 + wave;
    int colBase = ntile * 16;
    int col     = colBase + (lane & 15);

    for (int mtile = 0; mtile < 8; ++mtile) {
        v8f acc = {};
        for (int kb = 0; kb < K; kb += 32) {
            if (kb + 32 < K)  // pull next weight slab toward L2/L0
                __builtin_prefetch(Wn + (size_t)col * K + kb + 32, 0, 3);
            v16h a = load_a_frag_f16(logit, K, mtile * 16, kb, lane);
            v16h b = load_b_frag_f16(Wn, K, colBase, kb, lane);
            acc = __builtin_amdgcn_wmma_f32_16x16x32_f16(false, a, false, b,
                                                         (short)0, acc, false, false);
        }
        int rof = (lane < 16) ? 0 : 8;
#pragma unroll
        for (int r = 0; r < 8; ++r) {
            int row = mtile * 16 + r + rof;              // batch index
            int tgt = y[(t + 1) * BSZ + row];
            float v = (col == tgt) ? -3.0e38f : acc[r];
#pragma unroll
            for (int m = 1; m < 16; m <<= 1)             // max over 16 cols
                v = fmaxf(v, __shfl_xor(v, m, 32));
            if ((lane & 15) == 0)
                atomicMax(&mmax_key[row], fkey(v));
        }
    }
}

// ---------------------------------------------------------------------------
// Small elementwise / reduction kernels
// ---------------------------------------------------------------------------
__global__ void k_zero_out(float* out) { out[0] = 0.0f; }

__global__ void k_f32_to_f16(const float* __restrict__ src, _Float16* __restrict__ dst, int n) {
    int i = blockIdx.x * blockDim.x + threadIdx.x;
    if (i < n) dst[i] = (_Float16)src[i];
}

// normalize rows of W_out (V x 512) and write f16
__global__ void k_norm_wout(const float* __restrict__ W, _Float16* __restrict__ dst) {
    __shared__ float sh[256];
    int j = blockIdx.x, tid = threadIdx.x;
    const float* row = W + (size_t)j * D_IN;
    float v0 = row[tid], v1 = row[tid + 256];
    sh[tid] = v0 * v0 + v1 * v1;
    __syncthreads();
    for (int s = 128; s > 0; s >>= 1) {
        if (tid < s) sh[tid] += sh[tid + s];
        __syncthreads();
    }
    float rn = rsqrtf(sh[0]);
    _Float16* drow = dst + (size_t)j * D_IN;
    drow[tid]       = (_Float16)(v0 * rn);
    drow[tid + 256] = (_Float16)(v1 * rn);
}

__global__ void k_mean_ctx(const float* __restrict__ ctx, const float* __restrict__ mask,
                           _Float16* __restrict__ mean_h) {
    int idx = blockIdx.x * blockDim.x + threadIdx.x;   // B*H = 131072
    if (idx >= BSZ * HDIM) return;
    int b = idx >> 10, h = idx & 1023;
    float s = 0.f, ms = 0.f;
    for (int sIdx = 0; sIdx < SLEN; ++sIdx) {
        s  += ctx[((size_t)sIdx * BSZ + b) * HDIM + h];
        ms += mask[sIdx * BSZ + b];
    }
    mean_h[(size_t)b * HDIM + h] = (_Float16)(s / ms);
}

// h0 = tanh(pre + b_init): write f32 master + f16 GEMM copy
__global__ void k_tanh_bias(const float* __restrict__ in, const float* __restrict__ bias,
                            float* __restrict__ out, _Float16* __restrict__ out_h, int n) {
    int i = blockIdx.x * blockDim.x + threadIdx.x;
    if (i < n) {
        float v = tanhf(in[i] + bias[i & (HDIM - 1)]);
        out[i]   = v;
        out_h[i] = (_Float16)v;
    }
}

__global__ void k_gather_e(const float* __restrict__ emb, const int* __restrict__ y,
                           _Float16* __restrict__ e_h, int t) {
    int idx = blockIdx.x * blockDim.x + threadIdx.x;   // B*D_IN
    if (idx >= BSZ * D_IN) return;
    int b = idx >> 9, k = idx & 511;
    int tok = y[t * BSZ + b];
    e_h[idx] = (_Float16)emb[(size_t)tok * D_IN + k];
}

__global__ void k_gru_combine(const float* __restrict__ gi, const float* __restrict__ gh,
                              const float* __restrict__ bih, const float* __restrict__ bhh,
                              const float* __restrict__ hprev, float* __restrict__ hnew,
                              _Float16* __restrict__ hnew_h) {
    int idx = blockIdx.x * blockDim.x + threadIdx.x;   // B*H
    if (idx >= BSZ * HDIM) return;
    int b = idx >> 10, j = idx & 1023;
    size_t o = (size_t)b * 3 * HDIM + j;
    float r  = sigf(gi[o] + bih[j] + gh[o] + bhh[j]);
    float z  = sigf(gi[o + HDIM] + bih[j + HDIM] + gh[o + HDIM] + bhh[j + HDIM]);
    float nn = tanhf(gi[o + 2 * HDIM] + bih[j + 2 * HDIM]
                     + r * (gh[o + 2 * HDIM] + bhh[j + 2 * HDIM]));
    float hv = (1.0f - z) * nn + z * hprev[idx];
    hnew[idx]   = hv;
    hnew_h[idx] = (_Float16)hv;
}

// scores[s][b] = tanh(ctx_p[s,b,:] + q[b,:]) . v_att  + neg(s,b)
__global__ void k_scores(const float* __restrict__ ctx_p, const float* __restrict__ q,
                         const float* __restrict__ v_att, const float* __restrict__ mask,
                         float* __restrict__ scores) {
    __shared__ float sh[256];
    int sb = blockIdx.x, tid = threadIdx.x;            // sb in [0, S*B)
    int b = sb & (BSZ - 1);
    float p = 0.f;
    for (int h = tid; h < HDIM; h += 256)
        p += tanhf(ctx_p[(size_t)sb * HDIM + h] + q[(size_t)b * HDIM + h]) * v_att[h];
    sh[tid] = p;
    __syncthreads();
    for (int s = 128; s > 0; s >>= 1) {
        if (tid < s) sh[tid] += sh[tid + s];
        __syncthreads();
    }
    if (tid == 0)
        scores[sb] = sh[0] + (mask[sb] > 0.f ? 0.f : -1.0e9f);
}

__global__ void k_softmax(const float* __restrict__ scores, float* __restrict__ alpha) {
    __shared__ float sh[SLEN];
    int b = blockIdx.x, s = threadIdx.x;               // 64 threads
    float v = scores[s * BSZ + b];
    sh[s] = v;
    __syncthreads();
    for (int k = 32; k > 0; k >>= 1) {
        if (s < k) sh[s] = fmaxf(sh[s], sh[s + k]);
        __syncthreads();
    }
    float mx = sh[0];
    __syncthreads();
    float e = expf(v - mx);
    sh[s] = e;
    __syncthreads();
    for (int k = 32; k > 0; k >>= 1) {
        if (s < k) sh[s] += sh[s + k];
        __syncthreads();
    }
    alpha[s * BSZ + b] = e / sh[0];
}

// z_t (GEMM A operand only -> f16)
__global__ void k_ztr(const float* __restrict__ alpha, const float* __restrict__ ctx,
                      _Float16* __restrict__ z_h) {
    int idx = blockIdx.x * blockDim.x + threadIdx.x;   // B*H
    if (idx >= BSZ * HDIM) return;
    int b = idx >> 10, h = idx & 1023;
    float acc = 0.f;
    for (int s = 0; s < SLEN; ++s)
        acc += alpha[s * BSZ + b] * ctx[((size_t)s * BSZ + b) * HDIM + h];
    z_h[idx] = (_Float16)acc;
}

// tanh + bias + L2-normalize rows of lg (B x 512) -> f16; reset per-row max keys
__global__ void k_logit_norm(const float* __restrict__ lg, const float* __restrict__ bo,
                             _Float16* __restrict__ logit_h, unsigned* __restrict__ mmax_key) {
    __shared__ float sh[256];
    int b = blockIdx.x, tid = threadIdx.x;
    size_t base = (size_t)b * D_IN;
    float v0 = tanhf(lg[base + tid] + bo[tid]);
    float v1 = tanhf(lg[base + tid + 256] + bo[tid + 256]);
    sh[tid] = v0 * v0 + v1 * v1;
    __syncthreads();
    for (int s = 128; s > 0; s >>= 1) {
        if (tid < s) sh[tid] += sh[tid + s];
        __syncthreads();
    }
    float rn = rsqrtf(sh[0]);
    logit_h[base + tid]       = (_Float16)(v0 * rn);
    logit_h[base + tid + 256] = (_Float16)(v1 * rn);
    if (tid == 0) mmax_key[b] = fkey(-3.0e38f);
}

__global__ void k_strue(const _Float16* __restrict__ logit, const _Float16* __restrict__ Wn,
                        const int* __restrict__ y, float* __restrict__ s_true, int t) {
    __shared__ float sh[256];
    int b = blockIdx.x, tid = threadIdx.x;
    int tgt = y[(t + 1) * BSZ + b];
    float p = 0.f;
    for (int k = tid; k < D_IN; k += 256)
        p += (float)logit[(size_t)b * D_IN + k] * (float)Wn[(size_t)tgt * D_IN + k];
    sh[tid] = p;
    __syncthreads();
    for (int s = 128; s > 0; s >>= 1) {
        if (tid < s) sh[tid] += sh[tid + s];
        __syncthreads();
    }
    if (tid == 0) s_true[b] = sh[0];
}

__global__ void k_loss(const float* __restrict__ s_true, const unsigned* __restrict__ mmax_key,
                       const int* __restrict__ y, float* __restrict__ out, int t) {
    __shared__ float sh[BSZ];
    int b = threadIdx.x;
    int tgt = y[(t + 1) * BSZ + b];
    float l = 0.f;
    if (tgt != 0)
        l = fmaxf(0.f, MARGIN - s_true[b] + funkey(mmax_key[b]));
    sh[b] = l;
    __syncthreads();
    for (int s = 64; s > 0; s >>= 1) {
        if (b < s) sh[b] += sh[b + s];
        __syncthreads();
    }
    if (b == 0) out[0] += sh[0];
}

// ---------------------------------------------------------------------------
// Host launcher
// ---------------------------------------------------------------------------
extern "C" void kernel_launch(void* const* d_in, const int* in_sizes, int n_in,
                              void* d_out, int out_size, void* d_ws, size_t ws_size,
                              hipStream_t stream) {
    const float* ctx      = (const float*)d_in[0];
    const float* ctx_mask = (const float*)d_in[1];
    const int*   y        = (const int*)d_in[2];
    const float* emb_W    = (const float*)d_in[3];
    const float* W_init   = (const float*)d_in[4];
    const float* b_init   = (const float*)d_in[5];
    const float* W_ih0    = (const float*)d_in[6];
    const float* W_hh0    = (const float*)d_in[7];
    const float* b_ih0    = (const float*)d_in[8];
    const float* b_hh0    = (const float*)d_in[9];
    const float* W_c2c    = (const float*)d_in[10];
    const float* W_h2c    = (const float*)d_in[11];
    const float* v_att    = (const float*)d_in[12];
    const float* W_ih1    = (const float*)d_in[13];
    const float* W_hh1    = (const float*)d_in[14];
    const float* b_ih1    = (const float*)d_in[15];
    const float* b_hh1    = (const float*)d_in[16];
    const float* W_h2o    = (const float*)d_in[17];
    const float* b_h2o    = (const float*)d_in[18];
    const float* W_out    = (const float*)d_in[19];
    float* out = (float*)d_out;

    // ---- carve workspace (all regions 256B aligned) ----
    char* p = (char*)d_ws;
    auto carve = [&](size_t bytes) -> char* {
        char* r = p;
        p += (bytes + 255) & ~(size_t)255;
        return r;
    };
    _Float16* Wih0h  = (_Float16*)carve((size_t)3 * HDIM * D_IN * 2);
    _Float16* Whh0h  = (_Float16*)carve((size_t)3 * HDIM * HDIM * 2);
    _Float16* Wih1h  = (_Float16*)carve((size_t)3 * HDIM * HDIM * 2);
    _Float16* Whh1h  = (_Float16*)carve((size_t)3 * HDIM * HDIM * 2);
    _Float16* Wc2ch  = (_Float16*)carve((size_t)HDIM * HDIM * 2);
    _Float16* Wh2ch  = (_Float16*)carve((size_t)HDIM * HDIM * 2);
    _Float16* Wh2oh  = (_Float16*)carve((size_t)D_IN * HDIM * 2);
    _Float16* Winith = (_Float16*)carve((size_t)HDIM * HDIM * 2);
    _Float16* Woutnh = (_Float16*)carve((size_t)VOC * D_IN * 2);
    _Float16* ctx_h  = (_Float16*)carve((size_t)SLEN * BSZ * HDIM * 2);
    float* ctx_p  = (float*)carve((size_t)SLEN * BSZ * HDIM * 4);
    _Float16* mean_h = (_Float16*)carve((size_t)BSZ * HDIM * 2);
    float* hcur   = (float*)carve((size_t)BSZ * HDIM * 4);
    _Float16* hcur_h = (_Float16*)carve((size_t)BSZ * HDIM * 2);
    float* h1     = (float*)carve((size_t)BSZ * HDIM * 4);
    _Float16* h1_h = (_Float16*)carve((size_t)BSZ * HDIM * 2);
    float* q      = (float*)carve((size_t)BSZ * HDIM * 4);
    _Float16* z_h = (_Float16*)carve((size_t)BSZ * HDIM * 2);
    _Float16* e_h = (_Float16*)carve((size_t)BSZ * D_IN * 2);
    float* gi     = (float*)carve((size_t)BSZ * 3 * HDIM * 4);
    float* gh     = (float*)carve((size_t)BSZ * 3 * HDIM * 4);
    float* lg     = (float*)carve((size_t)BSZ * D_IN * 4);
    _Float16* logit_h = (_Float16*)carve((size_t)BSZ * D_IN * 2);
    float* scores = (float*)carve((size_t)SLEN * BSZ * 4);
    float* alpha  = (float*)carve((size_t)SLEN * BSZ * 4);
    float* s_true = (float*)carve((size_t)BSZ * 4);
    unsigned* mmax_key = (unsigned*)carve((size_t)BSZ * 4);

    auto cvt = [&](const float* src, _Float16* dst, int n) {
        k_f32_to_f16<<<(n + 255) / 256, 256, 0, stream>>>(src, dst, n);
    };

    // ---- one-time preprocessing ----
    k_zero_out<<<1, 1, 0, stream>>>(out);
    cvt(W_ih0, Wih0h, 3 * HDIM * D_IN);
    cvt(W_hh0, Whh0h, 3 * HDIM * HDIM);
    cvt(W_ih1, Wih1h, 3 * HDIM * HDIM);
    cvt(W_hh1, Whh1h, 3 * HDIM * HDIM);
    cvt(W_c2c, Wc2ch, HDIM * HDIM);
    cvt(W_h2c, Wh2ch, HDIM * HDIM);
    cvt(W_h2o, Wh2oh, D_IN * HDIM);
    cvt(W_init, Winith, HDIM * HDIM);
    cvt(ctx, ctx_h, SLEN * BSZ * HDIM);
    k_norm_wout<<<VOC, 256, 0, stream>>>(W_out, Woutnh);

    // h0 = tanh(mean_ctx @ W_init^T + b_init)
    k_mean_ctx<<<(BSZ * HDIM) / 256, 256, 0, stream>>>(ctx, ctx_mask, mean_h);
    k_gemm<<<dim3(HDIM / 64, BSZ / 16), 128, 0, stream>>>(mean_h, Winith, gi, BSZ, HDIM, HDIM);
    k_tanh_bias<<<(BSZ * HDIM) / 256, 256, 0, stream>>>(gi, b_init, hcur, hcur_h, BSZ * HDIM);

    // ctx_p = ctx @ W_c2c^T  (M = S*B = 8192)
    k_gemm<<<dim3(HDIM / 64, (SLEN * BSZ) / 16), 128, 0, stream>>>(ctx_h, Wc2ch, ctx_p,
                                                                   SLEN * BSZ, HDIM, HDIM);

    // ---- 47 decode steps ----
    for (int t = 0; t < TLEN - 1; ++t) {
        k_gather_e<<<(BSZ * D_IN) / 256, 256, 0, stream>>>(emb_W, y, e_h, t);
        // GRU0 gates
        k_gemm<<<dim3(3 * HDIM / 64, BSZ / 16), 128, 0, stream>>>(e_h, Wih0h, gi, BSZ, D_IN, 3 * HDIM);
        k_gemm<<<dim3(3 * HDIM / 64, BSZ / 16), 128, 0, stream>>>(hcur_h, Whh0h, gh, BSZ, HDIM, 3 * HDIM);
        k_gru_combine<<<(BSZ * HDIM) / 256, 256, 0, stream>>>(gi, gh, b_ih0, b_hh0, hcur, h1, h1_h);
        // attention
        k_gemm<<<dim3(HDIM / 64, BSZ / 16), 128, 0, stream>>>(h1_h, Wh2ch, q, BSZ, HDIM, HDIM);
        k_scores<<<SLEN * BSZ, 256, 0, stream>>>(ctx_p, q, v_att, ctx_mask, scores);
        k_softmax<<<BSZ, SLEN, 0, stream>>>(scores, alpha);
        k_ztr<<<(BSZ * HDIM) / 256, 256, 0, stream>>>(alpha, ctx, z_h);
        // GRU1 gates
        k_gemm<<<dim3(3 * HDIM / 64, BSZ / 16), 128, 0, stream>>>(z_h, Wih1h, gi, BSZ, HDIM, 3 * HDIM);
        k_gemm<<<dim3(3 * HDIM / 64, BSZ / 16), 128, 0, stream>>>(h1_h, Whh1h, gh, BSZ, HDIM, 3 * HDIM);
        k_gru_combine<<<(BSZ * HDIM) / 256, 256, 0, stream>>>(gi, gh, b_ih1, b_hh1, h1, hcur, hcur_h);
        // output projection + normalize
        k_gemm<<<dim3(D_IN / 64, BSZ / 16), 128, 0, stream>>>(hcur_h, Wh2oh, lg, BSZ, HDIM, D_IN);
        k_logit_norm<<<BSZ, 256, 0, stream>>>(lg, b_h2o, logit_h, mmax_key);
        // hinge loss pieces
        k_strue<<<BSZ, 256, 0, stream>>>(logit_h, Woutnh, y, s_true, t);
        k_vocab_max<<<(VOC / 16) / 4, 128, 0, stream>>>(logit_h, Woutnh, y, mmax_key, t);
        k_loss<<<1, BSZ, 0, stream>>>(s_true, mmax_key, y, out, t);
    }
}
